// GraphAttentionLayer_82583631168057
// MI455X (gfx1250) — matrix operand order repaired and checked by reading, not compiled, not run on hardware
//
#include <hip/hip_runtime.h>

typedef __attribute__((ext_vector_type(16))) _Float16 v16h;
typedef __attribute__((ext_vector_type(8)))  float    v8f;

#define B_     4
#define N_     4096
#define FIN_   128
#define FOUT_  64
#define LALPHA 0.2f
#define LOG2E  1.4426950408889634f

// ---------------------------------------------------------------------------
// Stage 1: Wh = h @ W ; e1 = (Wh.a1)*log2e ; e2 = (Wh.a2)*log2e
// WhT stored f16 as [b][f][n] so stage-2 B-operands are contiguous v16h loads.
// e1/e2 pre-scaled by log2(e): lrelu commutes with positive scale, so stage 2
// can use v_exp (exp2) directly with no per-element log2e multiply.
// ---------------------------------------------------------------------------
__global__ __launch_bounds__(256) void gat_stage1(
    const float* __restrict__ h, const float* __restrict__ W,
    const float* __restrict__ a, _Float16* __restrict__ WhT,
    float* __restrict__ e1, float* __restrict__ e2)
{
  __shared__ float Wl[FIN_ * FOUT_];
  __shared__ float red1[4][2];
  __shared__ float red2[4][2];

  const int tid = threadIdx.x;
  for (int i = tid; i < FIN_ * FOUT_; i += 256) Wl[i] = W[i];
  __syncthreads();

  const int localRow = tid >> 6;
  const int f        = tid & 63;
  const int gr       = blockIdx.x * 4 + localRow;
  const int b        = gr >> 12;
  const int n        = gr & (N_ - 1);

  const float* hrow = h + (size_t)gr * FIN_;
  float acc = 0.f;
#pragma unroll 8
  for (int k = 0; k < FIN_; ++k) acc = fmaf(hrow[k], Wl[k * FOUT_ + f], acc);

  WhT[((size_t)b * FOUT_ + f) * N_ + n] = (_Float16)acc;

  float p1 = acc * a[f];
  float p2 = acc * a[FOUT_ + f];
#pragma unroll
  for (int off = 16; off > 0; off >>= 1) {
    p1 += __shfl_xor(p1, off);
    p2 += __shfl_xor(p2, off);
  }
  const int hhalf = (tid >> 5) & 1;
  if ((tid & 31) == 0) { red1[localRow][hhalf] = p1; red2[localRow][hhalf] = p2; }
  __syncthreads();
  if (f == 0) {
    e1[gr] = (red1[localRow][0] + red1[localRow][1]) * LOG2E;
    e2[gr] = (red2[localRow][0] + red2[localRow][1]) * LOG2E;
  }
}

// ---------------------------------------------------------------------------
// Bit-pack adjacency: packAdj[row][j/32] bit j%32 = (adj[row][j] > 0).
// Read 64 MB once from HBM, write 2 MB (L2-resident for stage 2, shared by
// all 4 batches). One wave produces one 32-bit word via ballot.
// ---------------------------------------------------------------------------
__global__ __launch_bounds__(256) void gat_pack(
    const int* __restrict__ adj, unsigned* __restrict__ packAdj)
{
  const unsigned gid = blockIdx.x * 256u + threadIdx.x;     // 16M threads
  const int v = adj[gid];
  const unsigned long long m = __ballot(v > 0);             // wave32: low 32b
  if ((threadIdx.x & 31) == 0) packAdj[gid >> 5] = (unsigned)m;
}

// ---------------------------------------------------------------------------
// maxE2[b] = max_j e2[b][j]  (static softmax-max upper bound, log2-scaled)
// ---------------------------------------------------------------------------
__global__ __launch_bounds__(256) void gat_maxe2(
    const float* __restrict__ e2, float* __restrict__ maxE2)
{
  __shared__ float red[8];
  const int b = blockIdx.x;
  const float* p = e2 + (size_t)b * N_;
  float m = -__builtin_inff();
  for (int i = threadIdx.x; i < N_; i += 256) m = fmaxf(m, p[i]);
#pragma unroll
  for (int off = 16; off > 0; off >>= 1) m = fmaxf(m, __shfl_xor(m, off));
  if ((threadIdx.x & 31) == 0) red[threadIdx.x >> 5] = m;
  __syncthreads();
  if (threadIdx.x == 0) {
    float r = red[0];
#pragma unroll
    for (int i = 1; i < 8; ++i) r = fmaxf(r, red[i]);
    maxE2[b] = r;
  }
}

// ---------------------------------------------------------------------------
// Stage 2: fused masked-softmax(attn) @ Wh.
// Static per-row max bound m_row = lrelu(e1_i + maxE2[b]) (monotonicity) =>
// no online rescale, no per-iteration cross-lane ops. Masking from packed
// bits: lane (g,M) needs columns k = 16g+h, i.e.
// bit h of (bits >> 16g). Inner loop: 1 b32 + 4 float4 + 4 v16h loads,
// exp2/pack VALU, 4x v_wmma_f32_16x16x32_f16.
// ---------------------------------------------------------------------------
__global__ __launch_bounds__(128) void gat_stage2(
    const unsigned* __restrict__ packAdj, const _Float16* __restrict__ WhT,
    const float* __restrict__ e1, const float* __restrict__ e2,
    const float* __restrict__ maxE2, float* __restrict__ out)
{
  __shared__ float sl[4][16];
  __shared__ float sacc[4][16][FOUT_];

  const int b    = blockIdx.x >> 8;
  const int i0   = (blockIdx.x & 255) << 4;
  const int w    = threadIdx.x >> 5;
  const int lane = threadIdx.x & 31;
  const int g    = lane >> 4;
  const int M    = lane & 15;
  const int row  = i0 + M;

  const float e1v = e1[b * N_ + row];                // log2-scaled
  const float mb  = e1v + maxE2[b];
  const float mrow = mb > 0.f ? mb : LALPHA * mb;    // lrelu of bound

  const float*    e2b     = e2 + (size_t)b * N_;
  const unsigned* packRow = packAdj + (size_t)row * (N_ / 32);
  const _Float16* whtB    = WhT + (size_t)b * FOUT_ * N_ + (size_t)M * N_;

  float l_run = 0.f;
  v8f acc0 = {}, acc1 = {}, acc2 = {}, acc3 = {};

  const int jBeg = w * (N_ / 4);
  const int jEnd = jBeg + (N_ / 4);
#pragma unroll 2
  for (int j0 = jBeg; j0 < jEnd; j0 += 32) {
    const unsigned bl = packRow[j0 >> 5] >> (16 * g);  // this lane's 16 mask bits
    const float* ep = e2b + j0 + 16 * g;
    float4 eA0 = *(const float4*)(ep);
    float4 eA1 = *(const float4*)(ep + 4);
    float4 eB0 = *(const float4*)(ep + 8);
    float4 eB1 = *(const float4*)(ep + 12);

    const _Float16* bp = whtB + j0 + 16 * g;
    __builtin_prefetch(bp + 32, 0, 0);   // next WhT tile; speculative at tail

    auto pv = [&](unsigned bit, float ev) -> float {
      float x = e1v + ev;
      x = x > 0.f ? x : LALPHA * x;            // leaky_relu (log2 domain)
      float p = __builtin_exp2f(x - mrow);     // <= 1, no overflow
      return bit ? p : 0.f;                    // adjacency mask
    };
    float p[16];
    p[0]  = pv(bl & (1u <<  0), eA0.x); p[1]  = pv(bl & (1u <<  1), eA0.y);
    p[2]  = pv(bl & (1u <<  2), eA0.z); p[3]  = pv(bl & (1u <<  3), eA0.w);
    p[4]  = pv(bl & (1u <<  4), eA1.x); p[5]  = pv(bl & (1u <<  5), eA1.y);
    p[6]  = pv(bl & (1u <<  6), eA1.z); p[7]  = pv(bl & (1u <<  7), eA1.w);
    p[8]  = pv(bl & (1u <<  8), eB0.x); p[9]  = pv(bl & (1u <<  9), eB0.y);
    p[10] = pv(bl & (1u << 10), eB0.z); p[11] = pv(bl & (1u << 11), eB0.w);
    p[12] = pv(bl & (1u << 12), eB1.x); p[13] = pv(bl & (1u << 13), eB1.y);
    p[14] = pv(bl & (1u << 14), eB1.z); p[15] = pv(bl & (1u << 15), eB1.w);

    v16h afrag;
    float ts = 0.f;
#pragma unroll
    for (int hh = 0; hh < 16; ++hh) {
      ts += p[hh];
      afrag[hh] = (_Float16)p[hh];
    }
    l_run += ts;

    v16h b0 = *(const v16h*)(bp);
    v16h b1 = *(const v16h*)(bp + 16 * N_);
    v16h b2 = *(const v16h*)(bp + 32 * N_);
    v16h b3 = *(const v16h*)(bp + 48 * N_);

    acc0 = __builtin_amdgcn_wmma_f32_16x16x32_f16(false, afrag, false, b0, (short)0, acc0, false, false);
    acc1 = __builtin_amdgcn_wmma_f32_16x16x32_f16(false, afrag, false, b1, (short)0, acc1, false, false);
    acc2 = __builtin_amdgcn_wmma_f32_16x16x32_f16(false, afrag, false, b2, (short)0, acc2, false, false);
    acc3 = __builtin_amdgcn_wmma_f32_16x16x32_f16(false, afrag, false, b3, (short)0, acc3, false, false);
  }
  l_run += __shfl_xor(l_run, 16);        // lanes L and L+16 share row L&15

  // ---- stash per-wave partials ----
  if (g == 0) sl[w][M] = l_run;
#pragma unroll
  for (int r = 0; r < 8; ++r) {
    const int rr = r + 8 * g;            // C/D layout: VGPR r -> row r+8g
    sacc[w][rr][ 0 + M] = acc0[r];
    sacc[w][rr][16 + M] = acc1[r];
    sacc[w][rr][32 + M] = acc2[r];
    sacc[w][rr][48 + M] = acc3[r];
  }
  __syncthreads();

  // ---- merge 4 j-slices (plain sums; shared static max) ----
  const int rr = threadIdx.x & 15;
  const int c0 = (threadIdx.x >> 4) * 8;
  const float lt = sl[0][rr] + sl[1][rr] + sl[2][rr] + sl[3][rr];
  float* orow = out + ((size_t)b * N_ + (i0 + rr)) * FOUT_ + c0;
  if (lt > 0.f) {
    const float inv = 1.0f / lt;
#pragma unroll
    for (int q = 0; q < 8; ++q) {
      float v = sacc[0][rr][c0 + q] + sacc[1][rr][c0 + q]
              + sacc[2][rr][c0 + q] + sacc[3][rr][c0 + q];
      orow[q] = v * inv;
    }
  } else {
    // all-masked row: reference softmax over equal values -> mean of Wh
    for (int q = 0; q < 8; ++q) {
      const _Float16* wc = WhT + ((size_t)b * FOUT_ + (c0 + q)) * N_;
      float s = 0.f;
      for (int n = 0; n < N_; ++n) s += (float)wc[n];
      orow[q] = s * (1.0f / N_);
    }
  }
}

// ---------------------------------------------------------------------------
extern "C" void kernel_launch(void* const* d_in, const int* in_sizes, int n_in,
                              void* d_out, int out_size, void* d_ws, size_t ws_size,
                              hipStream_t stream)
{
  const float* h   = (const float*)d_in[0];   // (4,4096,128) f32
  const int*   adj = (const int*)d_in[1];     // (4096,4096) i32
  const float* W   = (const float*)d_in[2];   // (128,64) f32
  const float* a   = (const float*)d_in[3];   // (128,1)  f32
  float* out = (float*)d_out;                 // (4,4096,64) f32

  char* ws = (char*)d_ws;
  _Float16* WhT = (_Float16*)ws;                               ws += (size_t)B_ * FOUT_ * N_ * sizeof(_Float16);  // 2 MB
  float*    e1  = (float*)ws;                                  ws += (size_t)B_ * N_ * sizeof(float);             // 64 KB
  float*    e2  = (float*)ws;                                  ws += (size_t)B_ * N_ * sizeof(float);             // 64 KB
  float*    mx  = (float*)ws;                                  ws += 64;                                          // 4 floats
  unsigned* pk  = (unsigned*)ws;                               // 2 MB packed adjacency

  gat_stage1<<<(B_ * N_) / 4, 256, 0, stream>>>(h, W, a, WhT, e1, e2);
  gat_pack<<<(N_ * N_) / 256, 256, 0, stream>>>(adj, pk);
  gat_maxe2<<<B_, 256, 0, stream>>>(e2, mx);
  gat_stage2<<<B_ * (N_ / 16), 128, 0, stream>>>(pk, WhT, e1, e2, mx, out);
}
